// LlamaDecoderLayer_10763188043893
// MI455X (gfx1250) — matrix operand order repaired and checked by reading, not compiled, not run on hardware
//
#include <hip/hip_runtime.h>

// ---------------- problem constants (from reference) ----------------
#define H_     2048
#define NH     16
#define NKV    4
#define HD     128
#define INTER  8192
#define B_     4
#define T_     512
#define KVLEN  4096
#define CACHE_ 131072
#define BT     (B_ * T_)
#define EPS_   1e-6f

// gfx1250 async global->LDS staging (CDNA5 ISA). Flip to 0 if
// the assembler rejects the mnemonic.
#define USE_ASYNC_LDS 1

typedef __bf16 bf16;
typedef __attribute__((ext_vector_type(16))) bf16  v16bf;
typedef __attribute__((ext_vector_type(8)))  bf16  v8bf;
typedef __attribute__((ext_vector_type(8)))  float v8f;

__device__ __forceinline__ v16bf pack16(v8bf lo, v8bf hi) {
    v16bf r;
#pragma unroll
    for (int i = 0; i < 8; ++i) { r[i] = lo[i]; r[i + 8] = hi[i]; }
    return r;
}

__device__ __forceinline__ float redmax16(float x) {
#pragma unroll
    for (int off = 1; off < 16; off <<= 1) x = fmaxf(x, __shfl_xor(x, off, 32));
    return x;
}
__device__ __forceinline__ float redsum16(float x) {
#pragma unroll
    for (int off = 1; off < 16; off <<= 1) x += __shfl_xor(x, off, 32);
    return x;
}
__device__ __forceinline__ float redsum32(float x) {
#pragma unroll
    for (int off = 1; off < 32; off <<= 1) x += __shfl_xor(x, off, 32);
    return x;
}

#if USE_ASYNC_LDS
// Per-lane async copy of 16B global -> LDS. LDS address operand is a VGPR with
// the wave-relative LDS byte offset == low 32 bits of the generic pointer
// (LDS aperture lives in addr[63:32]). Tracked with ASYNCcnt.
__device__ __forceinline__ void async_ld_b128(const bf16* gsrc, void* ldst) {
    unsigned loff = (unsigned)(unsigned long long)ldst;
    asm volatile("global_load_async_to_lds_b128 %0, %1, off"
                 :: "v"(loff), "v"(gsrc)
                 : "memory");
}
__device__ __forceinline__ void wait_async0() {
    asm volatile("s_wait_asynccnt 0x0" ::: "memory");
}
#endif

// ---------------- small utility kernels ----------------
__global__ __launch_bounds__(256) void fill_i32_k(int* __restrict__ p, int val, int n) {
    int i = blockIdx.x * blockDim.x + threadIdx.x;
    if (i < n) p[i] = val;
}

__global__ __launch_bounds__(256) void build_inv_k(const int* __restrict__ lut,
                                                   int* __restrict__ inv, int n) {
    int i = blockIdx.x * blockDim.x + threadIdx.x;
    if (i < n) inv[lut[i]] = i;
}

// Fused f32->bf16 convert + transpose: out[n][k] = bf16(in[k][n]); in is [K][N].
// 32x32 LDS tile, coalesced on both sides. Grid: (N/32, K/32).
__global__ __launch_bounds__(256) void transpose_cvt_k(const float* __restrict__ in,
                                                       bf16* __restrict__ out,
                                                       int K, int N) {
    __shared__ bf16 tile[32][33];
    const int kt = blockIdx.y * 32, nt = blockIdx.x * 32;
    const int r = threadIdx.x >> 3, c = (threadIdx.x & 7) * 4;
    const float* src = in + (size_t)(kt + r) * N + nt + c;
#pragma unroll
    for (int i = 0; i < 4; ++i) tile[c + i][r] = (bf16)src[i];
    __syncthreads();
    bf16* dst = out + (size_t)(nt + r) * K + kt + c;
#pragma unroll
    for (int i = 0; i < 4; ++i) dst[i] = tile[r][c + i];
}

// bf16 transpose of gathered V: [S][HD] -> [HD][S], per (b,kv) slab.
// Grid: (HD/32, KVLEN/32, B*NKV).
__global__ __launch_bounds__(256) void transpose_v_k(const bf16* __restrict__ in,
                                                     bf16* __restrict__ out) {
    __shared__ bf16 tile[32][33];
    const size_t slab = (size_t)blockIdx.z * KVLEN * HD;
    const int st = blockIdx.y * 32, dt = blockIdx.x * 32;
    const int r = threadIdx.x >> 3, c = (threadIdx.x & 7) * 4;
    const bf16* src = in + slab + (size_t)(st + r) * HD + dt + c;
#pragma unroll
    for (int i = 0; i < 4; ++i) tile[c + i][r] = src[i];
    __syncthreads();
    bf16* dst = out + slab + (size_t)(dt + r) * KVLEN + st + c;
#pragma unroll
    for (int i = 0; i < 4; ++i) dst[i] = tile[r][c + i];
}

// rmsnorm: one block per row of [BT, H_]; writes bf16(w * x * rsqrt(mean(x^2)+eps))
__global__ __launch_bounds__(256) void rmsnorm_k(const float* __restrict__ x,
                                                 const float* __restrict__ w,
                                                 bf16* __restrict__ out) {
    __shared__ float red[8];
    const int row = blockIdx.x;
    const int tid = threadIdx.x, wave = tid >> 5, lane = tid & 31;
    const float* xr = x + (size_t)row * H_;
    float ss = 0.f;
    for (int i = tid; i < H_; i += 256) { float v = xr[i]; ss += v * v; }
    ss = redsum32(ss);
    if (lane == 0) red[wave] = ss;
    __syncthreads();
    float tot = 0.f;
#pragma unroll
    for (int i = 0; i < 8; ++i) tot += red[i];
    const float scale = rsqrtf(tot * (1.0f / H_) + EPS_);
    bf16* orow = out + (size_t)row * H_;
    for (int i = tid; i < H_; i += 256) orow[i] = (bf16)(xr[i] * scale * w[i]);
}

// silu(g) * u, bf16 in/out
__global__ __launch_bounds__(256) void silu_mul_k(const bf16* __restrict__ g,
                                                  const bf16* __restrict__ u,
                                                  bf16* __restrict__ out, long n) {
    long i = (long)blockIdx.x * blockDim.x + threadIdx.x;
    long stride = (long)gridDim.x * blockDim.x;
    for (; i < n; i += stride) {
        float gv = (float)g[i];
        float s = gv / (1.0f + __expf(-gv));
        out[i] = (bf16)(s * (float)u[i]);
    }
}

// RoPE on q/k, plus bf16 convert of v. One block per (b,t) row.
__global__ __launch_bounds__(256) void rope_k(const float* __restrict__ qf,
                                              const float* __restrict__ kf,
                                              const float* __restrict__ vf,
                                              const float* __restrict__ rope,
                                              const int* __restrict__ posoff,
                                              bf16* __restrict__ qb,
                                              bf16* __restrict__ kb,
                                              bf16* __restrict__ vb) {
    const int bt = blockIdx.x;
    const int b = bt / T_, t = bt % T_;
    const int tid = threadIdx.x;
    const float* rc = rope + (size_t)(posoff[b] + t) * HD;  // 64 (cos,sin) pairs
    for (int idx = tid; idx < NH * 64; idx += 256) {
        int head = idx >> 6, p = idx & 63;
        size_t base = (size_t)bt * (NH * HD) + (size_t)head * HD;
        float x1 = qf[base + p], x2 = qf[base + 64 + p];
        float c = rc[2 * p], s = rc[2 * p + 1];
        qb[base + p]      = (bf16)(x1 * c - x2 * s);
        qb[base + 64 + p] = (bf16)(x2 * c + x1 * s);
    }
    for (int idx = tid; idx < NKV * 64; idx += 256) {
        int head = idx >> 6, p = idx & 63;
        size_t base = (size_t)bt * (NKV * HD) + (size_t)head * HD;
        float x1 = kf[base + p], x2 = kf[base + 64 + p];
        float c = rc[2 * p], s = rc[2 * p + 1];
        kb[base + p]      = (bf16)(x1 * c - x2 * s);
        kb[base + 64 + p] = (bf16)(x2 * c + x1 * s);
    }
    for (int idx = tid; idx < NKV * HD; idx += 256) {
        size_t base = (size_t)bt * (NKV * HD);
        vb[base + idx] = (bf16)vf[base + idx];
    }
}

// Gather per-row KV sets via kv_lut, resolved against the inverse drain LUT so the
// 512MB zero cache is never cloned. Both K and V written row-major (coalesced);
// V is transposed afterwards by transpose_v_k.
__global__ __launch_bounds__(256) void kv_gather_k(const bf16* __restrict__ knew,
                                                   const bf16* __restrict__ vnew,
                                                   const float* __restrict__ kvcache,
                                                   const int* __restrict__ inv,
                                                   const int* __restrict__ kvlut,
                                                   bf16* __restrict__ kg,
                                                   bf16* __restrict__ vg) {
    long rid = (long)blockIdx.x * 2 + (threadIdx.x >> 7);
    int d = threadIdx.x & 127;
    int s = (int)(rid % KVLEN);
    int bk = (int)(rid / KVLEN);
    int kv = bk % NKV, b = bk / NKV;
    int slot = kvlut[(size_t)b * KVLEN + s];
    int j = inv[slot];
    bf16 kval, vval;
    if (j >= 0) {
        kval = knew[(size_t)j * (NKV * HD) + (size_t)kv * HD + d];
        vval = vnew[(size_t)j * (NKV * HD) + (size_t)kv * HD + d];
    } else {
        kval = (bf16)kvcache[((size_t)(0 * NKV + kv) * CACHE_ + slot) * HD + d];
        vval = (bf16)kvcache[((size_t)(1 * NKV + kv) * CACHE_ + slot) * HD + d];
    }
    size_t o = (((size_t)b * NKV + kv) * KVLEN + s) * HD + d;
    kg[o] = kval;
    vg[o] = vval;
}

// ---------------- bf16 WMMA GEMM: C[M,N] = A[M,K] @ W[K,N] (+resid) ----------------
// W is pre-transposed: Bt[N][K]. Both LDS tiles are contiguous-k, filled with
// async b128 global->LDS copies. Block = 8 waves, tile 128x128x32; each wave owns
// a 32x64 subtile: 2 A-frags x 4 B-frags -> 8 WMMAs per k-step, 12 LDS b128 reads.
#define TK  32
#define PAD 8

template <typename OutT, bool RESID>
__global__ __launch_bounds__(256) void gemm_bf16_wmma(const bf16* __restrict__ A,
                                                      const bf16* __restrict__ Bt,
                                                      const float* __restrict__ resid,
                                                      OutT* __restrict__ C,
                                                      int M, int N, int K) {
    __shared__ bf16 As[128][TK + PAD];
    __shared__ bf16 Bs[128][TK + PAD];
    const int tid = threadIdx.x, wave = tid >> 5, lane = tid & 31;
    const int half = lane >> 4, l16 = lane & 15;
    const int m0 = blockIdx.y * 128, n0 = blockIdx.x * 128;
    const int mstrip = (wave >> 1) * 32;  // 4 strips of 32 rows
    const int nstrip = (wave & 1) * 64;   // 2 strips of 64 cols

    v8f acc[2][4];
#pragma unroll
    for (int am = 0; am < 2; ++am)
#pragma unroll
        for (int j = 0; j < 4; ++j)
#pragma unroll
            for (int v = 0; v < 8; ++v) acc[am][j][v] = 0.f;

    const int r = tid >> 1, c0 = (tid & 1) * 16;  // 128 rows x 32 cols, 16 elem/thread
    for (int kk = 0; kk < K; kk += TK) {
        const bf16* ga = A + (size_t)(m0 + r) * K + kk + c0;
        const bf16* gb = Bt + (size_t)(n0 + r) * K + kk + c0;
#if USE_ASYNC_LDS
        async_ld_b128(ga,     &As[r][c0]);
        async_ld_b128(ga + 8, &As[r][c0 + 8]);
        async_ld_b128(gb,     &Bs[r][c0]);
        async_ld_b128(gb + 8, &Bs[r][c0 + 8]);
        wait_async0();           // drain this wave's ASYNCcnt before the barrier
#else
        *(v8bf*)&As[r][c0]     = *(const v8bf*)ga;
        *(v8bf*)&As[r][c0 + 8] = *(const v8bf*)(ga + 8);
        *(v8bf*)&Bs[r][c0]     = *(const v8bf*)gb;
        *(v8bf*)&Bs[r][c0 + 8] = *(const v8bf*)(gb + 8);
#endif
        __syncthreads();

        // A fragments (ISA 16-bit A 16x32 layout): e<8 -> k=8*half+e ; e>=8 -> +16
        v16bf af[2];
#pragma unroll
        for (int am = 0; am < 2; ++am) {
            int mrow = mstrip + am * 16 + l16;
            int kb = half * 8;
            v8bf lo = *(const v8bf*)&As[mrow][kb];
            v8bf hi = *(const v8bf*)&As[mrow][kb + 16];
            af[am] = pack16(lo, hi);
        }
#pragma unroll
        for (int j = 0; j < 4; ++j) {
            // B fragment (32x16): e<8 -> k=8*half+e ; e>=8 -> +16 (matches A)
            int ncol = nstrip + j * 16 + l16;
            int kb = half * 8;
            v8bf lo = *(const v8bf*)&Bs[ncol][kb];
            v8bf hi = *(const v8bf*)&Bs[ncol][kb + 16];
            v16bf bfrag = pack16(lo, hi);
#pragma unroll
            for (int am = 0; am < 2; ++am)
                acc[am][j] = __builtin_amdgcn_wmma_f32_16x16x32_bf16(
                    false, af[am], false, bfrag, (short)0, acc[am][j], false, false);
        }
        __syncthreads();
    }

    // Epilogue: C layout m = v + 8*half, n = l16
#pragma unroll
    for (int am = 0; am < 2; ++am) {
#pragma unroll
        for (int j = 0; j < 4; ++j) {
#pragma unroll
            for (int v = 0; v < 8; ++v) {
                int row = m0 + mstrip + am * 16 + v + 8 * half;
                int col = n0 + nstrip + j * 16 + l16;
                float val = acc[am][j][v];
                if (RESID) val += resid[(size_t)row * N + col];
                C[(size_t)row * N + col] = (OutT)val;
            }
        }
    }
}

// ---------------- flash attention, bf16 WMMA, online softmax ----------------
__global__ __launch_bounds__(256) void attn_k(const bf16* __restrict__ qb,   // [BT, NH*HD]
                                              const bf16* __restrict__ kg,   // [B][NKV][S][HD]
                                              const bf16* __restrict__ vgT,  // [B][NKV][HD][S]
                                              const unsigned char* __restrict__ mask, // [B][T][S]
                                              bf16* __restrict__ ob) {       // [BT, NH*HD]
    __shared__ bf16 Plds[8][16][TK + PAD];  // per-wave P scratch (16 x 32)
    const int tid = threadIdx.x, wave = tid >> 5, lane = tid & 31;
    const int half = lane >> 4, l16 = lane & 15;
    const int id = blockIdx.x * 8 + wave;
    const int ttile = id & 31;            // T/16 = 32
    const int h = (id >> 5) & 15;
    const int b = id >> 9;
    const int t0 = ttile * 16;
    const int kvh = h >> 2;               // NH/NKV = 4
    const bf16* Kb = kg + ((size_t)b * NKV + kvh) * (size_t)KVLEN * HD;
    const bf16* Vb = vgT + ((size_t)b * NKV + kvh) * (size_t)HD * KVLEN;
    const float sc = 0.08838834764831845f;  // 1/sqrt(128)

    v16bf qfrag[4];
    {
        size_t qrow = ((size_t)b * T_ + t0 + l16) * (NH * HD) + (size_t)h * HD;
#pragma unroll
        for (int c = 0; c < 4; ++c) {
            int d0 = c * 32 + half * 8;
            v8bf lo = *(const v8bf*)(qb + qrow + d0);
            v8bf hi = *(const v8bf*)(qb + qrow + d0 + 16);
            qfrag[c] = pack16(lo, hi);
        }
    }

    v8f acc[8];
    float mrow[8], lrow[8];
#pragma unroll
    for (int j = 0; j < 8; ++j)
#pragma unroll
        for (int v = 0; v < 8; ++v) acc[j][v] = 0.f;
#pragma unroll
    for (int v = 0; v < 8; ++v) { mrow[v] = -3.0e38f; lrow[v] = 0.f; }

    for (int s0 = 0; s0 < KVLEN; s0 += 32) {
        if (s0 + 32 < KVLEN) {  // global_prefetch_b8 for next chunk
            __builtin_prefetch(Kb + (size_t)(s0 + 32) * HD, 0, 0);
            __builtin_prefetch(Vb + (s0 + 32), 0, 0);
        }
        v8f S[2];
#pragma unroll
        for (int hs = 0; hs < 2; ++hs) {
#pragma unroll
            for (int v = 0; v < 8; ++v) S[hs][v] = 0.f;
            int scol = s0 + hs * 16;
#pragma unroll
            for (int c = 0; c < 4; ++c) {
                const bf16* kp = Kb + (size_t)(scol + l16) * HD + c * 32 + half * 8;
                v8bf lo = *(const v8bf*)kp;
                v8bf hi = *(const v8bf*)(kp + 16);
                v16bf kf = pack16(lo, hi);
                S[hs] = __builtin_amdgcn_wmma_f32_16x16x32_bf16(false, qfrag[c], false, kf,
                                                                (short)0, S[hs], false, false);
            }
        }
        float tmax[8];
#pragma unroll
        for (int v = 0; v < 8; ++v) {
            int t = t0 + v + 8 * half;
            const unsigned char* mp = mask + ((size_t)b * T_ + t) * KVLEN + s0;
            float x0 = mp[l16]      ? S[0][v] * sc : -3.0e38f;
            float x1 = mp[16 + l16] ? S[1][v] * sc : -3.0e38f;
            S[0][v] = x0; S[1][v] = x1;
            tmax[v] = fmaxf(x0, x1);
        }
#pragma unroll
        for (int v = 0; v < 8; ++v) tmax[v] = redmax16(tmax[v]);
        float corr[8], rs[8];
#pragma unroll
        for (int v = 0; v < 8; ++v) {
            float mn = fmaxf(mrow[v], tmax[v]);
            corr[v] = __expf(mrow[v] - mn);
            float e0 = __expf(S[0][v] - mn);
            float e1 = __expf(S[1][v] - mn);
            S[0][v] = e0; S[1][v] = e1;
            rs[v] = e0 + e1;
            mrow[v] = mn;
        }
#pragma unroll
        for (int v = 0; v < 8; ++v) rs[v] = redsum16(rs[v]);
#pragma unroll
        for (int v = 0; v < 8; ++v) lrow[v] = lrow[v] * corr[v] + rs[v];
#pragma unroll
        for (int j = 0; j < 8; ++j)
#pragma unroll
            for (int v = 0; v < 8; ++v) acc[j][v] *= corr[v];
#pragma unroll
        for (int v = 0; v < 8; ++v) {
            int rm = v + 8 * half;
            Plds[wave][rm][l16]      = (bf16)S[0][v];
            Plds[wave][rm][16 + l16] = (bf16)S[1][v];
        }
        asm volatile("s_wait_dscnt 0" ::: "memory");  // same-wave LDS RAW
        v16bf pf;
        {
            int kb = half * 8;
            v8bf lo = *(const v8bf*)&Plds[wave][l16][kb];
            v8bf hi = *(const v8bf*)&Plds[wave][l16][kb + 16];
            pf = pack16(lo, hi);
        }
#pragma unroll
        for (int j = 0; j < 8; ++j) {
            const bf16* vp = Vb + (size_t)(j * 16 + l16) * KVLEN + s0 + half * 8;
            v8bf lo = *(const v8bf*)vp;
            v8bf hi = *(const v8bf*)(vp + 16);
            v16bf vf = pack16(lo, hi);
            acc[j] = __builtin_amdgcn_wmma_f32_16x16x32_bf16(false, pf, false, vf,
                                                             (short)0, acc[j], false, false);
        }
    }
#pragma unroll
    for (int j = 0; j < 8; ++j) {
#pragma unroll
        for (int v = 0; v < 8; ++v) {
            float denom = lrow[v];
            float o = (denom > 0.f) ? acc[j][v] / denom : 0.f;
            size_t row = (size_t)b * T_ + t0 + v + 8 * half;
            ob[row * (NH * HD) + (size_t)h * HD + j * 16 + l16] = (bf16)o;
        }
    }
}

// ---------------- host launch ----------------
extern "C" void kernel_launch(void* const* d_in, const int* in_sizes, int n_in,
                              void* d_out, int out_size, void* d_ws, size_t ws_size,
                              hipStream_t stream) {
    const float* hidden  = (const float*)d_in[0];
    const float* kvcache = (const float*)d_in[1];
    const float* rope    = (const float*)d_in[2];
    const float* wq = (const float*)d_in[3];
    const float* wk = (const float*)d_in[4];
    const float* wv = (const float*)d_in[5];
    const float* wo = (const float*)d_in[6];
    const float* wg = (const float*)d_in[7];
    const float* wu = (const float*)d_in[8];
    const float* wd = (const float*)d_in[9];
    const float* ln1 = (const float*)d_in[10];
    const float* ln2 = (const float*)d_in[11];
    const int* posoff = (const int*)d_in[12];
    const int* drain  = (const int*)d_in[13];
    const int* kvlut  = (const int*)d_in[14];
    const unsigned char* mask = (const unsigned char*)d_in[15];  // jnp.bool_: 1 byte/elem
    float* out = (float*)d_out;

    char* base = (char*)d_ws;
    size_t off = 0;
    auto alloc = [&](size_t bytes) -> void* {
        void* p = base + off;
        off += (bytes + 255) & ~(size_t)255;
        return p;
    };
    // weights stored TRANSPOSED as [N][K] bf16
    bf16* wq_t = (bf16*)alloc((size_t)H_ * NH * HD * 2);
    bf16* wk_t = (bf16*)alloc((size_t)H_ * NKV * HD * 2);
    bf16* wv_t = (bf16*)alloc((size_t)H_ * NKV * HD * 2);
    bf16* wo_t = (bf16*)alloc((size_t)NH * HD * H_ * 2);
    bf16* wg_t = (bf16*)alloc((size_t)H_ * INTER * 2);
    bf16* wu_t = (bf16*)alloc((size_t)H_ * INTER * 2);
    bf16* wd_t = (bf16*)alloc((size_t)INTER * H_ * 2);
    bf16* hnorm = (bf16*)alloc((size_t)BT * H_ * 2);
    float* qf = (float*)alloc((size_t)BT * NH * HD * 4);
    float* kf = (float*)alloc((size_t)BT * NKV * HD * 4);
    float* vf = (float*)alloc((size_t)BT * NKV * HD * 4);
    bf16* qb   = (bf16*)alloc((size_t)BT * NH * HD * 2);
    bf16* knew = (bf16*)alloc((size_t)BT * NKV * HD * 2);
    bf16* vnew = (bf16*)alloc((size_t)BT * NKV * HD * 2);
    int*  inv  = (int*)alloc((size_t)CACHE_ * 4);
    bf16* kg   = (bf16*)alloc((size_t)B_ * NKV * KVLEN * HD * 2);
    bf16* vg   = (bf16*)alloc((size_t)B_ * NKV * KVLEN * HD * 2);
    bf16* vgT  = (bf16*)alloc((size_t)B_ * NKV * KVLEN * HD * 2);
    bf16* ob   = (bf16*)alloc((size_t)BT * NH * HD * 2);
    float* h2  = (float*)alloc((size_t)BT * H_ * 4);
    bf16* mb   = (bf16*)alloc((size_t)BT * H_ * 2);
    bf16* gb   = (bf16*)alloc((size_t)BT * INTER * 2);
    bf16* ub   = (bf16*)alloc((size_t)BT * INTER * 2);
    bf16* gub  = (bf16*)alloc((size_t)BT * INTER * 2);
    (void)ws_size; (void)n_in; (void)in_sizes; (void)out_size;

    auto tcvt = [&](const float* src, bf16* dst, int K, int N) {
        transpose_cvt_k<<<dim3(N / 32, K / 32), 256, 0, stream>>>(src, dst, K, N);
    };
    // 1) weights -> bf16, transposed [N][K] (once per launch)
    tcvt(wq, wq_t, H_, NH * HD);
    tcvt(wk, wk_t, H_, NKV * HD);
    tcvt(wv, wv_t, H_, NKV * HD);
    tcvt(wo, wo_t, NH * HD, H_);
    tcvt(wg, wg_t, H_, INTER);
    tcvt(wu, wu_t, H_, INTER);
    tcvt(wd, wd_t, INTER, H_);

    // 2) rmsnorm1
    rmsnorm_k<<<BT, 256, 0, stream>>>(hidden, ln1, hnorm);

    // 3) QKV GEMMs (fp32 out for RoPE precision)
    gemm_bf16_wmma<float, false><<<dim3((NH * HD) / 128, BT / 128), 256, 0, stream>>>(
        hnorm, wq_t, nullptr, qf, BT, NH * HD, H_);
    gemm_bf16_wmma<float, false><<<dim3((NKV * HD) / 128, BT / 128), 256, 0, stream>>>(
        hnorm, wk_t, nullptr, kf, BT, NKV * HD, H_);
    gemm_bf16_wmma<float, false><<<dim3((NKV * HD) / 128, BT / 128), 256, 0, stream>>>(
        hnorm, wv_t, nullptr, vf, BT, NKV * HD, H_);

    // 4) inverse drain LUT (avoids cloning the 512MB cache)
    fill_i32_k<<<CACHE_ / 256, 256, 0, stream>>>(inv, -1, CACHE_);
    build_inv_k<<<BT / 256, 256, 0, stream>>>(drain, inv, BT);

    // 5) RoPE q/k + bf16 convert
    rope_k<<<BT, 256, 0, stream>>>(qf, kf, vf, rope, posoff, qb, knew, vnew);

    // 6) KV gather (coalesced row-major), then tiled V transpose for WMMA B-frags
    kv_gather_k<<<(B_ * NKV * KVLEN) / 2, 256, 0, stream>>>(knew, vnew, kvcache, inv,
                                                            kvlut, kg, vg);
    transpose_v_k<<<dim3(HD / 32, KVLEN / 32, B_ * NKV), 256, 0, stream>>>(vg, vgT);

    // 7) flash attention
    attn_k<<<(B_ * NH * (T_ / 16)) / 8, 256, 0, stream>>>(qb, kg, vgT, mask, ob);

    // 8) output projection + residual -> h2
    gemm_bf16_wmma<float, true><<<dim3(H_ / 128, BT / 128), 256, 0, stream>>>(
        ob, wo_t, hidden, h2, BT, H_, NH * HD);

    // 9) rmsnorm2
    rmsnorm_k<<<BT, 256, 0, stream>>>(h2, ln2, mb);

    // 10) MLP: gate/up GEMMs (bf16 out), SwiGLU, down GEMM + residual -> d_out
    gemm_bf16_wmma<bf16, false><<<dim3(INTER / 128, BT / 128), 256, 0, stream>>>(
        mb, wg_t, nullptr, gb, BT, INTER, H_);
    gemm_bf16_wmma<bf16, false><<<dim3(INTER / 128, BT / 128), 256, 0, stream>>>(
        mb, wu_t, nullptr, ub, BT, INTER, H_);
    silu_mul_k<<<8192, 256, 0, stream>>>(gb, ub, gub, (long)BT * INTER);
    gemm_bf16_wmma<float, true><<<dim3(H_ / 128, BT / 128), 256, 0, stream>>>(
        gub, wd_t, h2, out, BT, H_, INTER);
}
